// MarchMadnessModel_v6_74148315398276
// MI455X (gfx1250) — compile-verified, matrix operand order checked by reading, not run on hardware
//
#include <hip/hip_runtime.h>
#include <stdint.h>

// ---------------------------------------------------------------------------
// MI455X / gfx1250 implementation. All dense GEMMs use v_wmma_f32_16x16x32_bf16
// (wave32, f32 accumulate). Softmax / LN / BN / SiLU epilogues stay f32.
// ---------------------------------------------------------------------------

#define EPSF 1e-5f

typedef __bf16 bf16;
typedef __attribute__((ext_vector_type(16))) __bf16 v16bf;
typedef __attribute__((ext_vector_type(8)))  float  v8f;

__device__ __forceinline__ v8f wmma_bf16(v16bf a, v16bf b, v8f c) {
  // (neg_a, A, neg_b, B, c_mod, C, reuse_a, reuse_b)
  return __builtin_amdgcn_wmma_f32_16x16x32_bf16(false, a, false, b, (short)0, c,
                                                 false, false);
}

// 16x32 bf16 fragment (A tile, or B as weight-rows W[n][k] with n = lane%16).
// Layout per CDNA5 ISA 7.12.2: lanes 0-15 / 16-31 hold the same 16 rows with
// K split into two contiguous 8-element runs selected by lane[4].
__device__ __forceinline__ v16bf load_frag(const bf16* base, int row0, int stride,
                                           int k0) {
  int lane = threadIdx.x & 31;
  const bf16* p = base + (size_t)(row0 + (lane & 15)) * stride + k0 +
                  ((lane & 16) ? 8 : 0);
  v16bf f;
#pragma unroll
  for (int i = 0; i < 8; ++i) f[i] = p[i];
#pragma unroll
  for (int i = 0; i < 8; ++i) f[8 + i] = p[16 + i];
  return f;
}

__device__ __forceinline__ float silu_f(float x) { return x / (1.f + __expf(-x)); }

// ---------------------------------------------------------------------------
// K0: weight conversion / repack kernels (run once per launch, tiny)
// ---------------------------------------------------------------------------
__global__ void cvt_kernel(const float* __restrict__ src, bf16* __restrict__ dst,
                           int n) {
  int i = blockIdx.x * blockDim.x + threadIdx.x;
  if (i < n) dst[i] = (bf16)src[i];
}

// conv_w (64,128,3) -> convbf[h][t*128+k] so conv1d == GEMM over padded rows
__global__ void pack_conv_kernel(const float* __restrict__ src,
                                 bf16* __restrict__ dst) {
  int i = blockIdx.x * blockDim.x + threadIdx.x;
  if (i >= 64 * 384) return;
  int h = i / 384, r = i % 384;
  int t = r >> 7, k = r & 127;
  dst[i] = (bf16)src[(h * 128 + k) * 3 + t];
}

// tw_w (128,257) -> twbf[128][256] skipping elo column 128; twcol = column 128
__global__ void pack_tw_kernel(const float* __restrict__ src,
                               bf16* __restrict__ dst, float* __restrict__ col) {
  int i = blockIdx.x * blockDim.x + threadIdx.x;
  if (i >= 128 * 256) return;
  int h = i >> 8, c = i & 255;
  dst[i] = (bf16)src[h * 257 + c + (c >= 128 ? 1 : 0)];
  if (c == 0) col[h] = src[h * 257 + 128];
}

// ---------------------------------------------------------------------------
// K1: per-matchup history encoder + 4-way attention (1 block = 1 matchup)
// ---------------------------------------------------------------------------
struct P1 {
  const int *idA, *idB, *seedA, *seedB;
  const float *eloA, *eloB, *hnA, *hnB;
  const int *oppA, *oppB;
  const float *maskA, *maskB;
  const float *team_emb, *qp_w, *qp_b, *attn_in_w, *attn_in_b;
  const float *attn_out_w, *attn_out_b, *ln_g, *ln_b;
  const float *hp_b, *conv_b, *cbn_g, *cbn_b, *cbn_rm, *cbn_rv;
  const bf16 *embbf, *seedbf, *hpbf, *convbf, *wkvbf;
  bf16* feat;   // [16384][256] packed tower features (ea|sa|attn1|attn2)
  float* za;    // [16384] elo z-score
};

__global__ __launch_bounds__(128) void encode_attend_kernel(P1 p) {
  const int b = blockIdx.x;
  const int tid = threadIdx.x;
  const int wave = tid >> 5, lane = tid & 31;

  // xpad: [2][52][128] bf16 rows 1..40 = masked x, rows 0/41.. zero (conv pad).
  // Reused after phase 2 as the bf16 K|V output buffer [2][40][128].
  __shared__ __align__(16) bf16 s_x[2 * 52 * 128];
  __shared__ __align__(16) bf16 s_kv[2 * 48 * 64];  // kv = silu(bn(conv)+res)
  __shared__ float s_q[4][64], s_qh[4][64], s_sc[4][4][40];
  __shared__ float s_o[4][64], s_y[4][64], s_mx[4][4], s_is[4][4];

  const int idA = p.idA[b], idB = p.idB[b];
  const int sdA = p.seedA[b], sdB = p.seedB[b];

  __builtin_prefetch(p.hnA + (size_t)b * 1280, 0, 1);
  __builtin_prefetch(p.hnB + (size_t)b * 1280, 0, 1);

  // feat base columns: [0:96]=team emb, [96:128]=seed emb ; za scalar
  if (tid < 96) {
    p.feat[(size_t)(2 * b) * 256 + tid]     = p.embbf[idA * 96 + tid];
    p.feat[(size_t)(2 * b + 1) * 256 + tid] = p.embbf[idB * 96 + tid];
  } else {
    int j = tid - 96;
    p.feat[(size_t)(2 * b) * 256 + 96 + j]     = p.seedbf[sdA * 32 + j];
    p.feat[(size_t)(2 * b + 1) * 256 + 96 + j] = p.seedbf[sdB * 32 + j];
  }
  if (tid == 0) p.za[2 * b]     = (p.eloA[b] - 1500.f) * (1.f / 400.f);
  if (tid == 1) p.za[2 * b + 1] = (p.eloB[b] - 1500.f) * (1.f / 400.f);

  for (int i = tid; i < 2 * 52 * 128; i += 128) s_x[i] = (bf16)0.f;
  for (int i = tid; i < 2 * 48 * 64; i += 128)  s_kv[i] = (bf16)0.f;
  __syncthreads();

  // masked input rows: x[l] = [hn[l]*m , team_emb[opp[l]]*m]
  for (int i = tid; i < 2 * 40 * 128; i += 128) {
    int s = i / (40 * 128);
    int rem = i % (40 * 128);
    int l = rem >> 7, k = rem & 127;
    float m = (s ? p.maskB : p.maskA)[b * 40 + l];
    float v;
    if (k < 32) {
      v = (s ? p.hnB : p.hnA)[((size_t)b * 40 + l) * 32 + k];
    } else {
      int opp = (s ? p.oppB : p.oppA)[b * 40 + l];
      v = (float)p.embbf[opp * 96 + (k - 32)];
    }
    s_x[(s * 52 + l + 1) * 128 + k] = (bf16)(v * m);
  }
  __syncthreads();

  // Phase 2: res = silu(x@hp_w.T+b) ; conv-as-GEMM (K=384) ; kv = silu(bn(c)+res)
  for (int t = wave; t < 24; t += 4) {          // 2 sides x (3 mtiles x 4 ntiles)
    int s = t / 12, tt = t % 12;
    int mt = tt >> 2, nt = tt & 3;
    const bf16* xs = s_x + s * 52 * 128;
    v8f acc_hp = {};
#pragma unroll
    for (int kk = 0; kk < 4; ++kk) {
      v16bf a = load_frag(xs + 128, mt * 16, 128, kk * 32);   // row l -> xpad l+1
      v16bf w = load_frag(p.hpbf, nt * 16, 128, kk * 32);
      acc_hp = wmma_bf16(a, w, acc_hp);
    }
    v8f acc_cv = {};
#pragma unroll
    for (int kk = 0; kk < 12; ++kk) {
      v16bf a = load_frag(xs, mt * 16, 128, kk * 32);  // flattened xpad[l..l+2]
      v16bf w = load_frag(p.convbf, nt * 16, 384, kk * 32);
      acc_cv = wmma_bf16(a, w, acc_cv);
    }
    int n = nt * 16 + (lane & 15);
    float hb = p.hp_b[n], cb = p.conv_b[n];
    float scale = p.cbn_g[n] * rsqrtf(p.cbn_rv[n] + EPSF);
    float shift = p.cbn_b[n] - p.cbn_rm[n] * scale;
    int mb = mt * 16 + ((lane & 16) ? 8 : 0);
#pragma unroll
    for (int r = 0; r < 8; ++r) {
      int l = mb + r;
      if (l < 40) {
        float res = silu_f(acc_hp[r] + hb);
        float cc = (acc_cv[r] + cb) * scale + shift;
        s_kv[(s * 48 + l) * 64 + n] = (bf16)silu_f(cc + res);
      }
    }
  }
  __syncthreads();

  // Phase 3: K|V projection, [40,64] x [64,128] per side (reuse s_x as output)
  bf16* kvo = s_x;                                   // [2][40][128]: 0:64 K, 64:128 V
  for (int t = wave; t < 48; t += 4) {               // 2 x (3 mtiles x 8 ntiles)
    int s = t / 24, tt = t % 24;
    int mt = tt >> 3, nt = tt & 7;
    v8f acc = {};
#pragma unroll
    for (int kk = 0; kk < 2; ++kk) {
      v16bf a = load_frag(s_kv + s * 48 * 64, mt * 16, 64, kk * 32);
      v16bf w = load_frag(p.wkvbf, nt * 16, 64, kk * 32);
      acc = wmma_bf16(a, w, acc);
    }
    int n = nt * 16 + (lane & 15);
    float bias = p.attn_in_b[64 + n];                // bk then bv, contiguous
    int mb = mt * 16 + ((lane & 16) ? 8 : 0);
#pragma unroll
    for (int r = 0; r < 8; ++r) {
      int l = mb + r;
      if (l < 40) kvo[(s * 40 + l) * 128 + n] = (bf16)(acc[r] + bias);
    }
  }
  __syncthreads();

  // Phase 4: one wave per attention combo: aa, ab, bb, ba (KV shared per side)
  const int c = wave;
  const int side = (c == 0 || c == 3) ? 0 : 1;
  const int qid = (c < 2) ? idA : idB;
  const float* qe = p.team_emb + (size_t)qid * 96;

  for (int h = lane; h < 64; h += 32) {              // q = q_emb @ qp_w.T + qp_b
    float a = p.qp_b[h];
    const float* wr = p.qp_w + h * 96;
    for (int k = 0; k < 96; ++k) a += qe[k] * wr[k];
    s_q[c][h] = a;
  }
  __syncthreads();
  for (int h = lane; h < 64; h += 32) {              // qh = q @ wq.T + bq
    float a = p.attn_in_b[h];
    const float* wr = p.attn_in_w + h * 64;
    for (int k = 0; k < 64; ++k) a += s_q[c][k] * wr[k];
    s_qh[c][h] = a;
  }
  __syncthreads();
  const float* maskp = side ? p.maskB : p.maskA;
  const bf16* kvs = kvo + side * 40 * 128;
  for (int i = lane; i < 160; i += 32) {             // scores [4 heads][40]
    int h = i / 40, l = i % 40;
    float a = 0.f;
    for (int d = 0; d < 16; ++d)
      a += s_qh[c][h * 16 + d] * (float)kvs[l * 128 + h * 16 + d];
    a *= 0.25f;                                      // 1/sqrt(16)
    if (maskp[b * 40 + l] <= 0.f) a = -1e9f;
    s_sc[c][h][l] = a;
  }
  __syncthreads();
  if (lane < 4) {                                    // per-head softmax stats
    float mx = -3.4e38f;
    for (int l = 0; l < 40; ++l) mx = fmaxf(mx, s_sc[c][lane][l]);
    float sum = 0.f;
    for (int l = 0; l < 40; ++l) sum += __expf(s_sc[c][lane][l] - mx);
    s_mx[c][lane] = mx;
    s_is[c][lane] = 1.f / sum;
  }
  __syncthreads();
  for (int hd = lane; hd < 64; hd += 32) {           // o = softmax @ V
    int h = hd >> 4;
    float mx = s_mx[c][h], is = s_is[c][h];
    float a = 0.f;
    for (int l = 0; l < 40; ++l)
      a += __expf(s_sc[c][h][l] - mx) * is * (float)kvs[l * 128 + 64 + hd];
    s_o[c][hd] = a;
  }
  __syncthreads();
  for (int j = lane; j < 64; j += 32) {              // out proj + residual q
    float a = p.attn_out_b[j] + s_q[c][j];
    const float* wr = p.attn_out_w + j * 64;
    for (int k = 0; k < 64; ++k) a += s_o[c][k] * wr[k];
    s_y[c][j] = a;
  }
  __syncthreads();
  float mean = 0.f;                                  // layernorm over 64
  for (int k = 0; k < 64; ++k) mean += s_y[c][k];
  mean *= (1.f / 64.f);
  float var = 0.f;
  for (int k = 0; k < 64; ++k) { float d = s_y[c][k] - mean; var += d * d; }
  var *= (1.f / 64.f);
  float inv = rsqrtf(var + EPSF);
  int dstrow = (c < 2) ? (2 * b) : (2 * b + 1);
  int dstcol = (c == 0 || c == 2) ? 128 : 192;
  for (int j = lane; j < 64; j += 32) {
    float v = (s_y[c][j] - mean) * inv * p.ln_g[j] + p.ln_b[j];
    p.feat[(size_t)dstrow * 256 + dstcol + j] = (bf16)v;
  }
}

// ---------------------------------------------------------------------------
// K2: tower + merged residual MLP + heads (1 block = 16 matchups, 8 waves)
// ---------------------------------------------------------------------------
struct P2 {
  const bf16 *feat, *twbf, *r1bf, *r2bf;
  const float *twcol, *za;
  const float *tw_b, *tbn_g, *tbn_b, *tbn_rm, *tbn_rv;
  const float *r1_b, *r1bn_g, *r1bn_b, *r1bn_rm, *r1bn_rv;
  const float *r2_b, *r2bn_g, *r2bn_b, *r2bn_rm, *r2bn_rv;
  const float *mu_w, *mu_b, *lv_w, *lv_b, *wl_w, *wl_b;
  float* out;
};

__global__ __launch_bounds__(256) void tower_head_kernel(P2 p) {
  const int bg = blockIdx.x;                 // 16 matchups -> 32 feature rows
  const int tid = threadIdx.x, wave = tid >> 5, lane = tid & 31;

  __shared__ float s_ra[32 * 128];           // silu(bn(tower)) for ta/tb rows
  __shared__ float s_xf[16 * 256];           // merged residual state (f32)
  __shared__ __align__(16) bf16 s_xb[16 * 256];

  const bf16* featg = p.feat + (size_t)32 * bg * 256;

  // Tower: [32,256] x [256,128] + rank-1 elo column + BN + SiLU
  for (int t = wave; t < 16; t += 8) {       // 2 mtiles x 8 ntiles
    int mt = t >> 3, nt = t & 7;
    v8f acc = {};
#pragma unroll
    for (int kk = 0; kk < 8; ++kk) {
      v16bf a = load_frag(featg, mt * 16, 256, kk * 32);
      v16bf w = load_frag(p.twbf, nt * 16, 256, kk * 32);
      acc = wmma_bf16(a, w, acc);
    }
    int n = nt * 16 + (lane & 15);
    float tb = p.tw_b[n], tc = p.twcol[n];
    float scale = p.tbn_g[n] * rsqrtf(p.tbn_rv[n] + EPSF);
    float shift = p.tbn_b[n] - p.tbn_rm[n] * scale;
    int mb = mt * 16 + ((lane & 16) ? 8 : 0);
#pragma unroll
    for (int r = 0; r < 8; ++r) {
      int m = mb + r;
      float v = acc[r] + tb + p.za[32 * bg + m] * tc;
      s_ra[m * 128 + n] = silu_f(v * scale + shift);
    }
  }
  __syncthreads();

  // merged x = [ra - rb , ra * rb]
  for (int i = tid; i < 16 * 128; i += 256) {
    int bi = i >> 7, n = i & 127;
    float ra = s_ra[(2 * bi) * 128 + n], rb = s_ra[(2 * bi + 1) * 128 + n];
    float d = ra - rb, pr = ra * rb;
    s_xf[bi * 256 + n] = d;        s_xb[bi * 256 + n] = (bf16)d;
    s_xf[bi * 256 + 128 + n] = pr; s_xb[bi * 256 + 128 + n] = (bf16)pr;
  }
  __syncthreads();

  // Two residual blocks: x = silu(bn(x@W.T + b)) + x   ([16,256] x [256,256])
  for (int blk = 0; blk < 2; ++blk) {
    const bf16* wbm = blk ? p.r2bf : p.r1bf;
    const float* bb = blk ? p.r2_b : p.r1_b;
    const float* g  = blk ? p.r2bn_g : p.r1bn_g;
    const float* be = blk ? p.r2bn_b : p.r1bn_b;
    const float* rm = blk ? p.r2bn_rm : p.r1bn_rm;
    const float* rv = blk ? p.r2bn_rv : p.r1bn_rv;
    for (int nt = wave; nt < 16; nt += 8) {
      v8f acc = {};
#pragma unroll
      for (int kk = 0; kk < 8; ++kk) {
        v16bf a = load_frag(s_xb, 0, 256, kk * 32);
        v16bf w = load_frag(wbm, nt * 16, 256, kk * 32);
        acc = wmma_bf16(a, w, acc);
      }
      int n = nt * 16 + (lane & 15);
      float scale = g[n] * rsqrtf(rv[n] + EPSF);
      float shift = be[n] - rm[n] * scale;
      float bias = bb[n];
      int mb = (lane & 16) ? 8 : 0;
#pragma unroll
      for (int r = 0; r < 8; ++r) {
        int m = mb + r;
        s_xf[m * 256 + n] += silu_f((acc[r] + bias) * scale + shift);
      }
    }
    __syncthreads();
    for (int i = tid; i < 16 * 256; i += 256) s_xb[i] = (bf16)s_xf[i];
    __syncthreads();
  }

  // Heads: box_mu [B,8] | box_log_var [B,8] | win_logit [B]
  for (int t = tid; t < 16 * 17; t += 256) {
    int i = t / 17, j = t % 17;
    int bglob = 16 * bg + i;
    const float* w; float bias; float* dst;
    if (j < 8)       { w = p.mu_w + j * 256;       bias = p.mu_b[j];
                       dst = p.out + (size_t)bglob * 8 + j; }
    else if (j < 16) { w = p.lv_w + (j - 8) * 256; bias = p.lv_b[j - 8];
                       dst = p.out + 65536 + (size_t)bglob * 8 + (j - 8); }
    else             { w = p.wl_w;                 bias = p.wl_b[0];
                       dst = p.out + 131072 + bglob; }
    float a = bias;
    for (int k = 0; k < 256; ++k) a += s_xf[i * 256 + k] * w[k];
    *dst = a;
  }
}

// ---------------------------------------------------------------------------
// Host launcher
// ---------------------------------------------------------------------------
extern "C" void kernel_launch(void* const* d_in, const int* in_sizes, int n_in,
                              void* d_out, int out_size, void* d_ws,
                              size_t ws_size, hipStream_t stream) {
  (void)in_sizes; (void)n_in; (void)out_size; (void)ws_size;
  char* ws = (char*)d_ws;
  size_t off = 0;
  auto take = [&](size_t bytes) -> char* {
    char* pp = ws + off;
    off = (off + bytes + 255) & ~(size_t)255;
    return pp;
  };
  bf16* embbf  = (bf16*)take(400 * 96 * 2);
  bf16* seedbf = (bf16*)take(17 * 32 * 2);
  bf16* hpbf   = (bf16*)take(64 * 128 * 2);
  bf16* convbf = (bf16*)take(64 * 384 * 2);
  bf16* wkvbf  = (bf16*)take(128 * 64 * 2);
  bf16* twbf   = (bf16*)take(128 * 256 * 2);
  float* twcol = (float*)take(128 * 4);
  bf16* r1bf   = (bf16*)take(256 * 256 * 2);
  bf16* r2bf   = (bf16*)take(256 * 256 * 2);
  bf16* feat   = (bf16*)take((size_t)16384 * 256 * 2);
  float* za    = (float*)take(16384 * 4);

  cvt_kernel<<<150, 256, 0, stream>>>((const float*)d_in[12], embbf, 38400);
  cvt_kernel<<<3, 256, 0, stream>>>((const float*)d_in[13], seedbf, 544);
  cvt_kernel<<<32, 256, 0, stream>>>((const float*)d_in[20], hpbf, 8192);
  pack_conv_kernel<<<96, 256, 0, stream>>>((const float*)d_in[14], convbf);
  cvt_kernel<<<32, 256, 0, stream>>>((const float*)d_in[24] + 64 * 64, wkvbf, 8192);
  pack_tw_kernel<<<128, 256, 0, stream>>>((const float*)d_in[30], twbf, twcol);
  cvt_kernel<<<256, 256, 0, stream>>>((const float*)d_in[36], r1bf, 65536);
  cvt_kernel<<<256, 256, 0, stream>>>((const float*)d_in[42], r2bf, 65536);

  P1 p1;
  p1.idA = (const int*)d_in[0];     p1.idB = (const int*)d_in[1];
  p1.seedA = (const int*)d_in[2];   p1.seedB = (const int*)d_in[3];
  p1.eloA = (const float*)d_in[4];  p1.eloB = (const float*)d_in[5];
  p1.hnA = (const float*)d_in[6];   p1.hnB = (const float*)d_in[7];
  p1.oppA = (const int*)d_in[8];    p1.oppB = (const int*)d_in[9];
  p1.maskA = (const float*)d_in[10]; p1.maskB = (const float*)d_in[11];
  p1.team_emb = (const float*)d_in[12];
  p1.qp_w = (const float*)d_in[22]; p1.qp_b = (const float*)d_in[23];
  p1.attn_in_w = (const float*)d_in[24]; p1.attn_in_b = (const float*)d_in[25];
  p1.attn_out_w = (const float*)d_in[26]; p1.attn_out_b = (const float*)d_in[27];
  p1.ln_g = (const float*)d_in[28]; p1.ln_b = (const float*)d_in[29];
  p1.hp_b = (const float*)d_in[21]; p1.conv_b = (const float*)d_in[15];
  p1.cbn_g = (const float*)d_in[16]; p1.cbn_b = (const float*)d_in[17];
  p1.cbn_rm = (const float*)d_in[18]; p1.cbn_rv = (const float*)d_in[19];
  p1.embbf = embbf; p1.seedbf = seedbf; p1.hpbf = hpbf;
  p1.convbf = convbf; p1.wkvbf = wkvbf;
  p1.feat = feat; p1.za = za;
  encode_attend_kernel<<<8192, 128, 0, stream>>>(p1);

  P2 p2;
  p2.feat = feat; p2.twbf = twbf; p2.r1bf = r1bf; p2.r2bf = r2bf;
  p2.twcol = twcol; p2.za = za;
  p2.tw_b = (const float*)d_in[31];
  p2.tbn_g = (const float*)d_in[32]; p2.tbn_b = (const float*)d_in[33];
  p2.tbn_rm = (const float*)d_in[34]; p2.tbn_rv = (const float*)d_in[35];
  p2.r1_b = (const float*)d_in[37];
  p2.r1bn_g = (const float*)d_in[38]; p2.r1bn_b = (const float*)d_in[39];
  p2.r1bn_rm = (const float*)d_in[40]; p2.r1bn_rv = (const float*)d_in[41];
  p2.r2_b = (const float*)d_in[43];
  p2.r2bn_g = (const float*)d_in[44]; p2.r2bn_b = (const float*)d_in[45];
  p2.r2bn_rm = (const float*)d_in[46]; p2.r2bn_rv = (const float*)d_in[47];
  p2.mu_w = (const float*)d_in[48]; p2.mu_b = (const float*)d_in[49];
  p2.lv_w = (const float*)d_in[50]; p2.lv_b = (const float*)d_in[51];
  p2.wl_w = (const float*)d_in[52]; p2.wl_b = (const float*)d_in[53];
  p2.out = (float*)d_out;
  tower_head_kernel<<<512, 256, 0, stream>>>(p2);
}
